// NonLocalNetwork_44186623541796
// MI455X (gfx1250) — matrix operand order repaired and checked by reading, not compiled, not run on hardware
//
#include <hip/hip_runtime.h>
#include <math.h>

#define BATCH 4
#define NPTS  2048
#define CCH   128
#define HCH   64
#define NLAYER 6
#define EPSV  1e-5f

// ---------- CDNA5 WMMA types ----------
typedef __attribute__((ext_vector_type(16))) __bf16 bf16x16;
typedef __attribute__((ext_vector_type(8)))  float  v8f;
typedef __attribute__((ext_vector_type(4)))  unsigned int u32x4;

union B16x16 { bf16x16 v; u32x4 q[2]; unsigned short u[16]; };

__device__ inline unsigned short f2bf(float f) {
    unsigned int u = __builtin_bit_cast(unsigned int, f);
    unsigned int r = u + 0x7FFFu + ((u >> 16) & 1u);   // round-to-nearest-even
    return (unsigned short)(r >> 16);
}

__device__ inline v8f wmma_bf16(const B16x16& a, const B16x16& b, v8f c) {
    return __builtin_amdgcn_wmma_f32_16x16x32_bf16(false, a.v, false, b.v,
                                                   (short)0, c, false, false);
}

// ---------- small helpers ----------
__global__ void k_cvt_bf16(const float* __restrict__ in, unsigned short* __restrict__ out, int n) {
    int i = blockIdx.x * blockDim.x + threadIdx.x;
    if (i < n) out[i] = f2bf(in[i]);
}

// x = w0[128x6] * cors[B,6,N] + b0   (Cin=6: tiny, plain VALU)
__global__ void k_conv0(const float* __restrict__ xin, const float* __restrict__ w,
                        const float* __restrict__ b, float* __restrict__ xout) {
    int g = blockIdx.x * blockDim.x + threadIdx.x;
    if (g >= BATCH * CCH * NPTS) return;
    int n = g % NPTS;
    int o = (g / NPTS) % CCH;
    int bb = g / (NPTS * CCH);
    float s = b[o];
#pragma unroll
    for (int i = 0; i < 6; ++i)
        s += w[o * 6 + i] * xin[((size_t)(bb * 6 + i)) * NPTS + n];
    xout[g] = s;
}

// ---------- generic 1x1-conv GEMM: Y[o,n] = sum_k W[o,k]*X[k,n] + bias[o] ----------
// grid.x = BATCH*(NPTS/16); blockDim = (COUT/16)*32. Whole B panel (CIN x 16) staged
// to LDS once, then CIN/32 back-to-back WMMAs per wave.
// MODE 0: f32 out [B,COUT,N]; MODE 1: bf16 out [B,N,COUT]; MODE 2: bf16 out [B,COUT,N]
template <int CIN, int COUT, int MODE>
__global__ void k_conv_wmma(const float* __restrict__ X, const unsigned short* __restrict__ W,
                            const float* __restrict__ bias, void* __restrict__ out) {
    constexpr int NTHR = (COUT / 16) * 32;
    __shared__ __align__(16) unsigned short ldsB[CIN * 16];
    int nt = blockIdx.x % (NPTS / 16);
    int bb = blockIdx.x / (NPTS / 16);
    int n0 = nt * 16;
    int tid = threadIdx.x, lane = tid & 31, wave = tid >> 5;
    int o_base = wave * 16;
    const unsigned short* Wrow = W + (size_t)(o_base + (lane & 15)) * CIN;
    int kh = (lane >> 4) * 16;   // A-layout K sub-offset

    // stage full B panel: X[0..CIN) x [n0..n0+16) -> LDS in B layout (per 32-k chunk)
#pragma unroll
    for (int e = tid; e < CIN * 16; e += NTHR) {
        int kg = e >> 4, nl = e & 15;
        float xv = X[((size_t)(bb * CIN + kg)) * NPTS + n0 + nl];
        int chunk = kg >> 5, kl = kg & 31;
        int L = ((kl >> 4) << 4) + nl;
        ldsB[chunk * 512 + L * 16 + (kl & 15)] = f2bf(xv);
    }
    __syncthreads();

    v8f acc; for (int r = 0; r < 8; ++r) acc[r] = 0.f;
#pragma unroll
    for (int c = 0; c < CIN / 32; ++c) {
        B16x16 a, bm;
        a.q[0]  = *(const u32x4*)(Wrow + c * 32 + kh);
        a.q[1]  = *(const u32x4*)(Wrow + c * 32 + kh + 8);
        bm.q[0] = *(const u32x4*)(&ldsB[c * 512 + lane * 16]);
        bm.q[1] = *(const u32x4*)(&ldsB[c * 512 + lane * 16 + 8]);
        acc = wmma_bf16(a, bm, acc);
    }

    int nn = n0 + (lane & 15);
    int og = o_base + ((lane >> 4) << 3);
#pragma unroll
    for (int r = 0; r < 8; ++r) {
        float val = acc[r] + bias[og + r];
        if (MODE == 0)
            ((float*)out)[((size_t)(bb * COUT + og + r)) * NPTS + nn] = val;
        else if (MODE == 1)
            ((unsigned short*)out)[((size_t)(bb * NPTS + nn)) * COUT + og + r] = f2bf(val);
        else
            ((unsigned short*)out)[((size_t)(bb * COUT + og + r)) * NPTS + nn] = f2bf(val);
    }
}

// ---------- S[m,n] = (Q^T K)[m,n] * beta[m,n] ----------
// One wave computes a 16(m) x 64(n) strip: 4 accumulators share the A (Q) operand.
// Waves = BATCH * 128 * 32; blockDim = 256 (8 waves/block).
__global__ void k_qk(const unsigned short* __restrict__ Qb, const unsigned short* __restrict__ Kb,
                     const float* __restrict__ beta, float* __restrict__ S) {
    int wid = blockIdx.x * (blockDim.x >> 5) + (threadIdx.x >> 5);
    int lane = threadIdx.x & 31;
    const int NSTRIP = NPTS / 64;            // 32 strips of 64 n
    int ns = wid % NSTRIP;
    int mt = (wid / NSTRIP) % (NPTS / 16);
    int bb = wid / (NSTRIP * (NPTS / 16));
    int m0 = mt * 16, n0 = ns * 64;
    size_t rowbase = (size_t)bb * NPTS * NPTS;
    __builtin_prefetch(beta + rowbase + (size_t)m0 * NPTS + n0, 0, 0);

    const unsigned short* qrow = Qb + ((size_t)(bb * NPTS) + m0 + (lane & 15)) * CCH;
    const unsigned short* krow = Kb + ((size_t)(bb * NPTS) + n0 + (lane & 15)) * CCH;
    int kh  = (lane >> 4) * 8;
    int kh2 = (lane >> 4) * 16;
    v8f acc[4];
#pragma unroll
    for (int t = 0; t < 4; ++t)
        for (int r = 0; r < 8; ++r) acc[t][r] = 0.f;

#pragma unroll
    for (int c = 0; c < CCH; c += 32) {
        B16x16 a;
        a.q[0] = *(const u32x4*)(qrow + c + kh);
        a.q[1] = *(const u32x4*)(qrow + c + kh + 16);
#pragma unroll
        for (int t = 0; t < 4; ++t) {
            B16x16 bm;
            bm.q[0] = *(const u32x4*)(krow + (size_t)t * 16 * CCH + c + kh);
            bm.q[1] = *(const u32x4*)(krow + (size_t)t * 16 * CCH + c + kh + 16);
            acc[t] = wmma_bf16(a, bm, acc[t]);
        }
    }

    int mg = m0 + ((lane >> 4) << 3);
#pragma unroll
    for (int t = 0; t < 4; ++t) {
        int nn = n0 + t * 16 + (lane & 15);
#pragma unroll
        for (int r = 0; r < 8; ++r) {
            size_t idx = rowbase + (size_t)(mg + r) * NPTS + nn;
            S[idx] = acc[t][r] * beta[idx];
        }
    }
}

// ---------- per-row max and 1/sum(exp) ----------
__global__ void k_rowstats(const float* __restrict__ S, float* __restrict__ rmax,
                           float* __restrict__ rrcp) {
    __shared__ float red[256];
    int row = blockIdx.x;
    const float* p = S + (size_t)row * NPTS;
    int tid = threadIdx.x;
    float mx = -3.4e38f;
    for (int n = tid; n < NPTS; n += 256) mx = fmaxf(mx, p[n]);
    red[tid] = mx; __syncthreads();
    for (int s = 128; s > 0; s >>= 1) { if (tid < s) red[tid] = fmaxf(red[tid], red[tid + s]); __syncthreads(); }
    mx = red[0]; __syncthreads();
    float sum = 0.f;
    for (int n = tid; n < NPTS; n += 256) sum += __expf(p[n] - mx);
    red[tid] = sum; __syncthreads();
    for (int s = 128; s > 0; s >>= 1) { if (tid < s) red[tid] += red[tid + s]; __syncthreads(); }
    if (tid == 0) { rmax[row] = mx; rrcp[row] = 1.f / red[0]; }
}

// ---------- msg[c,n] = sum_m V[c,m] * exp(S[m,n]-max[m])*rrcp[m] ----------
// grid.x = BATCH*(NPTS/16); 256 threads (8 waves -> 128 channels x 16 points).
// 128 rows of P staged per barrier pair -> 4 WMMAs between barriers.
__global__ void k_pv(const unsigned short* __restrict__ Vb, const float* __restrict__ S,
                     const float* __restrict__ rmax, const float* __restrict__ rrcp,
                     float* __restrict__ out) {
    __shared__ __align__(16) unsigned short ldsB[128 * 16];
    int nt = blockIdx.x % (NPTS / 16);
    int bb = blockIdx.x / (NPTS / 16);
    int n0 = nt * 16;
    int tid = threadIdx.x, lane = tid & 31, wave = tid >> 5;
    int c_base = wave * 16;
    const unsigned short* vrow = Vb + ((size_t)(bb * CCH) + c_base + (lane & 15)) * NPTS;
    int kh = (lane >> 4) * 16;
    size_t Sbase = (size_t)bb * NPTS * NPTS;
    const float* rmx = rmax + bb * NPTS;
    const float* rrc = rrcp + bb * NPTS;
    v8f acc; for (int r = 0; r < 8; ++r) acc[r] = 0.f;

    for (int m0 = 0; m0 < NPTS; m0 += 128) {
        __syncthreads();
#pragma unroll
        for (int e = tid; e < 128 * 16; e += 256) {
            int mg = e >> 4, nl = e & 15;
            int m = m0 + mg;
            float s  = S[Sbase + (size_t)m * NPTS + n0 + nl];
            float ev = __expf(s - rmx[m]) * rrc[m];
            int chunk = mg >> 5, ml = mg & 31;
            int L = ((ml >> 4) << 4) + nl;
            ldsB[chunk * 512 + L * 16 + (ml & 15)] = f2bf(ev);
        }
        __syncthreads();
#pragma unroll
        for (int c = 0; c < 4; ++c) {
            B16x16 a, bm;
            a.q[0]  = *(const u32x4*)(vrow + m0 + c * 32 + kh);
            a.q[1]  = *(const u32x4*)(vrow + m0 + c * 32 + kh + 8);
            bm.q[0] = *(const u32x4*)(&ldsB[c * 512 + lane * 16]);
            bm.q[1] = *(const u32x4*)(&ldsB[c * 512 + lane * 16 + 8]);
            acc = wmma_bf16(a, bm, acc);
        }
    }
    int nn = n0 + (lane & 15);
    int cg = c_base + ((lane >> 4) << 3);
#pragma unroll
    for (int r = 0; r < 8; ++r)
        out[((size_t)(bb * CCH + cg + r)) * NPTS + nn] = acc[r];
}

// ---------- BatchNorm (training-mode) ----------
__global__ void k_bn_stats(const float* __restrict__ Y, int Ch, float* __restrict__ mean,
                           float* __restrict__ var) {
    __shared__ float rs[256], rs2[256];
    int c = blockIdx.x, tid = threadIdx.x;
    float s = 0.f, s2 = 0.f;
    for (int bb = 0; bb < BATCH; ++bb) {
        const float* p = Y + ((size_t)(bb * Ch + c)) * NPTS;
        for (int n = tid; n < NPTS; n += 256) { float v = p[n]; s += v; s2 += v * v; }
    }
    rs[tid] = s; rs2[tid] = s2; __syncthreads();
    for (int st = 128; st > 0; st >>= 1) {
        if (tid < st) { rs[tid] += rs[tid + st]; rs2[tid] += rs2[tid + st]; }
        __syncthreads();
    }
    if (tid == 0) {
        float cnt = (float)(BATCH * NPTS);
        float m = rs[0] / cnt;
        mean[c] = m;
        var[c] = rs2[0] / cnt - m * m;
    }
}

__global__ void k_bn_relu(const float* __restrict__ Y, int Ch, const float* __restrict__ mean,
                          const float* __restrict__ var, const float* __restrict__ g,
                          const float* __restrict__ be, float* __restrict__ Xo) {
    int i = blockIdx.x * blockDim.x + threadIdx.x;
    if (i >= BATCH * Ch * NPTS) return;
    int c = (i / NPTS) % Ch;
    float v = (Y[i] - mean[c]) * rsqrtf(var[c] + EPSV) * g[c] + be[c];
    Xo[i] = fmaxf(v, 0.f);
}

__global__ void k_add(float* __restrict__ x, const float* __restrict__ y, int n) {
    int i = blockIdx.x * blockDim.x + threadIdx.x;
    if (i < n) x[i] += y[i];
}

__global__ void k_copy(const float* __restrict__ x, float* __restrict__ o, int n) {
    int i = blockIdx.x * blockDim.x + threadIdx.x;
    if (i < n) o[i] = x[i];
}

// ============================= host side =============================
extern "C" void kernel_launch(void* const* d_in, const int* in_sizes, int n_in,
                              void* d_out, int out_size, void* d_ws, size_t ws_size,
                              hipStream_t stream) {
    const float* cors  = (const float*)d_in[0];
    const float* beta  = (const float*)d_in[1];
    const float* w0    = (const float*)d_in[2];
    const float* b0    = (const float*)d_in[3];
    const float* pc_w  = (const float*)d_in[4];
    const float* pc_b  = (const float*)d_in[5];
    const float* pc_g  = (const float*)d_in[6];
    const float* pc_be = (const float*)d_in[7];
    const float* q_w   = (const float*)d_in[8];
    const float* q_b   = (const float*)d_in[9];
    const float* k_w   = (const float*)d_in[10];
    const float* k_b   = (const float*)d_in[11];
    const float* v_w   = (const float*)d_in[12];
    const float* v_b   = (const float*)d_in[13];
    const float* m1_w  = (const float*)d_in[14];
    const float* m1_b  = (const float*)d_in[15];
    const float* m1_g  = (const float*)d_in[16];
    const float* m1_be = (const float*)d_in[17];
    const float* m2_w  = (const float*)d_in[18];
    const float* m2_b  = (const float*)d_in[19];
    const float* m2_g  = (const float*)d_in[20];
    const float* m2_be = (const float*)d_in[21];
    const float* m3_w  = (const float*)d_in[22];
    const float* m3_b  = (const float*)d_in[23];

    char* wsb = (char*)d_ws;
    size_t off = 0;
    auto alloc = [&](size_t bytes) -> void* {
        void* p = wsb + off;
        off = (off + bytes + 255) & ~(size_t)255;
        return p;
    };

    float* S    = (float*)alloc((size_t)BATCH * NPTS * NPTS * 4);   // 64 MB, L2-resident
    float* x    = (float*)alloc((size_t)BATCH * CCH * NPTS * 4);
    float* y    = (float*)alloc((size_t)BATCH * CCH * NPTS * 4);
    float* msg  = (float*)alloc((size_t)BATCH * CCH * NPTS * 4);
    float* h1   = (float*)alloc((size_t)BATCH * HCH * NPTS * 4);
    float* h2   = (float*)alloc((size_t)BATCH * HCH * NPTS * 4);
    unsigned short* Qb = (unsigned short*)alloc((size_t)BATCH * NPTS * CCH * 2);
    unsigned short* Kb = (unsigned short*)alloc((size_t)BATCH * NPTS * CCH * 2);
    unsigned short* Vb = (unsigned short*)alloc((size_t)BATCH * NPTS * CCH * 2);
    float* rmaxb = (float*)alloc((size_t)BATCH * NPTS * 4);
    float* rrcpb = (float*)alloc((size_t)BATCH * NPTS * 4);
    float* meanb = (float*)alloc(CCH * 4);
    float* varb  = (float*)alloc(CCH * 4);
    unsigned short* pc_wb = (unsigned short*)alloc((size_t)NLAYER * CCH * CCH * 2);
    unsigned short* q_wb  = (unsigned short*)alloc((size_t)NLAYER * CCH * CCH * 2);
    unsigned short* k_wb  = (unsigned short*)alloc((size_t)NLAYER * CCH * CCH * 2);
    unsigned short* v_wb  = (unsigned short*)alloc((size_t)NLAYER * CCH * CCH * 2);
    unsigned short* m1_wb = (unsigned short*)alloc((size_t)NLAYER * HCH * CCH * 2);
    unsigned short* m2_wb = (unsigned short*)alloc((size_t)NLAYER * HCH * HCH * 2);
    unsigned short* m3_wb = (unsigned short*)alloc((size_t)NLAYER * CCH * HCH * 2);
    if (off > ws_size) return;  // workspace too small: bail (deterministic)

    auto cvt = [&](const float* src, unsigned short* dst, int n) {
        k_cvt_bf16<<<(n + 255) / 256, 256, 0, stream>>>(src, dst, n);
    };
    cvt(pc_w, pc_wb, NLAYER * CCH * CCH);
    cvt(q_w,  q_wb,  NLAYER * CCH * CCH);
    cvt(k_w,  k_wb,  NLAYER * CCH * CCH);
    cvt(v_w,  v_wb,  NLAYER * CCH * CCH);
    cvt(m1_w, m1_wb, NLAYER * HCH * CCH);
    cvt(m2_w, m2_wb, NLAYER * HCH * HCH);
    cvt(m3_w, m3_wb, NLAYER * CCH * HCH);

    const int tot128 = BATCH * CCH * NPTS;   // 1048576
    const int tot64  = BATCH * HCH * NPTS;   // 524288
    const int convGrid = BATCH * (NPTS / 16);                       // 512
    const int qkGrid = (BATCH * (NPTS / 16) * (NPTS / 64)) / 8;     // 2048

    k_conv0<<<(tot128 + 255) / 256, 256, 0, stream>>>(cors, w0, b0, x);

    for (int i = 0; i < NLAYER; ++i) {
        // --- PointCN: conv + BN + ReLU ---
        k_conv_wmma<CCH, CCH, 0><<<convGrid, 256, 0, stream>>>(x, pc_wb + (size_t)i * CCH * CCH,
                                                               pc_b + i * CCH, y);
        k_bn_stats<<<CCH, 256, 0, stream>>>(y, CCH, meanb, varb);
        k_bn_relu<<<(tot128 + 255) / 256, 256, 0, stream>>>(y, CCH, meanb, varb,
                                                            pc_g + i * CCH, pc_be + i * CCH, x);
        // --- Q, K, V projections (bf16, WMMA-ready layouts) ---
        k_conv_wmma<CCH, CCH, 1><<<convGrid, 256, 0, stream>>>(x, q_wb + (size_t)i * CCH * CCH,
                                                               q_b + i * CCH, Qb);
        k_conv_wmma<CCH, CCH, 1><<<convGrid, 256, 0, stream>>>(x, k_wb + (size_t)i * CCH * CCH,
                                                               k_b + i * CCH, Kb);
        k_conv_wmma<CCH, CCH, 2><<<convGrid, 256, 0, stream>>>(x, v_wb + (size_t)i * CCH * CCH,
                                                               v_b + i * CCH, Vb);
        // --- attention ---
        k_qk<<<qkGrid, 256, 0, stream>>>(Qb, Kb, beta, S);
        k_rowstats<<<BATCH * NPTS, 256, 0, stream>>>(S, rmaxb, rrcpb);
        k_pv<<<convGrid, 256, 0, stream>>>(Vb, S, rmaxb, rrcpb, msg);
        // --- MLP: conv-BN-ReLU x2, conv ---
        k_conv_wmma<CCH, HCH, 0><<<convGrid, 128, 0, stream>>>(msg, m1_wb + (size_t)i * HCH * CCH,
                                                               m1_b + i * HCH, h1);
        k_bn_stats<<<HCH, 256, 0, stream>>>(h1, HCH, meanb, varb);
        k_bn_relu<<<(tot64 + 255) / 256, 256, 0, stream>>>(h1, HCH, meanb, varb,
                                                           m1_g + i * HCH, m1_be + i * HCH, h1);
        k_conv_wmma<HCH, HCH, 0><<<convGrid, 128, 0, stream>>>(h1, m2_wb + (size_t)i * HCH * HCH,
                                                               m2_b + i * HCH, h2);
        k_bn_stats<<<HCH, 256, 0, stream>>>(h2, HCH, meanb, varb);
        k_bn_relu<<<(tot64 + 255) / 256, 256, 0, stream>>>(h2, HCH, meanb, varb,
                                                           m2_g + i * HCH, m2_be + i * HCH, h2);
        k_conv_wmma<HCH, CCH, 0><<<convGrid, 256, 0, stream>>>(h2, m3_wb + (size_t)i * CCH * HCH,
                                                               m3_b + i * CCH, y);
        // --- residual ---
        k_add<<<(tot128 + 255) / 256, 256, 0, stream>>>(x, y, tot128);
    }
    k_copy<<<(tot128 + 255) / 256, 256, 0, stream>>>(x, (float*)d_out, tot128);
}
